// GQA_49331994362471
// MI455X (gfx1250) — matrix operand order repaired and checked by reading, not compiled, not run on hardware
//
#include <hip/hip_runtime.h>

// ---------------------------------------------------------------------------
// Types for CDNA5 WMMA (gfx1250, wave32)
// ---------------------------------------------------------------------------
typedef __attribute__((ext_vector_type(16))) __bf16        v16bf;
typedef __attribute__((ext_vector_type(8)))  float         v8f;
typedef __attribute__((ext_vector_type(4)))  unsigned int  u32x4;
typedef __attribute__((ext_vector_type(4)))  int           v4i;

union FragBF {
    v16bf v;
    u32x4 q[2];
};

__device__ __forceinline__ v8f wmma_bf16(v16bf a, v16bf b, v8f c) {
    // D = A(16x32) * B(32x16) + C, fp32 accumulate
    return __builtin_amdgcn_wmma_f32_16x16x32_bf16(
        /*neg_a=*/false, a, /*neg_b=*/false, b,
        /*c_mod=*/(short)0, c, /*reuse_a=*/false, /*reuse_b=*/false);
}

// ---- CDNA5 async global->LDS copy (ASYNCcnt-tracked), with safe fallback ----
__device__ __forceinline__ void cp_async16(const __bf16* g, __bf16* l) {
#if __has_builtin(__builtin_amdgcn_global_load_async_to_lds_b128)
    // builtin signature: (v4i AS1* gsrc, v4i AS3* ldst, imm offset, imm cpol)
    __builtin_amdgcn_global_load_async_to_lds_b128(
        (__attribute__((address_space(1))) v4i*)g,
        (__attribute__((address_space(3))) v4i*)l,
        /*offset=*/0, /*cpol=*/0);
#else
    *(u32x4*)l = *(const u32x4*)g;   // sync fallback: global_load + ds_store
#endif
}

__device__ __forceinline__ void wait_async0() {
#if __has_builtin(__builtin_amdgcn_global_load_async_to_lds_b128)
#if __has_builtin(__builtin_amdgcn_s_wait_asynccnt)
    __builtin_amdgcn_s_wait_asynccnt(0);
#else
    asm volatile("s_wait_asynccnt 0" ::: "memory");
#endif
#endif
    // fallback path: ds_store tracked by DScnt; __syncthreads() covers it
}

__device__ __forceinline__ float hmax16(float v) {
    // reduce across the 16-lane half-wave (xor masks stay within halves)
    v = fmaxf(v, __shfl_xor(v, 1, 32));
    v = fmaxf(v, __shfl_xor(v, 2, 32));
    v = fmaxf(v, __shfl_xor(v, 4, 32));
    v = fmaxf(v, __shfl_xor(v, 8, 32));
    return v;
}
__device__ __forceinline__ float hsum16(float v) {
    v += __shfl_xor(v, 1, 32);
    v += __shfl_xor(v, 2, 32);
    v += __shfl_xor(v, 4, 32);
    v += __shfl_xor(v, 8, 32);
    return v;
}

// ---------------------------------------------------------------------------
// Elementwise conversions / layout kernels
// ---------------------------------------------------------------------------
__global__ void k_f32_to_bf16(const float* __restrict__ in,
                              __bf16* __restrict__ out, int n) {
    int i = blockIdx.x * 256 + threadIdx.x;
    if (i < n) out[i] = (__bf16)in[i];
}

// in: R x C fp32 row-major  ->  out: C x R bf16 row-major (transposed)
__global__ void k_transpose_bf16(const float* __restrict__ in,
                                 __bf16* __restrict__ out, int R, int C) {
    int i = blockIdx.x * 256 + threadIdx.x;
    if (i < R * C) {
        int r = i / C, c = i % C;
        out[(size_t)c * R + r] = (__bf16)in[(size_t)r * C + c];
    }
}

// Vf: [4096][1024] fp32  ->  Vt: [b][kvh][128][2048] bf16 (d-major per head)
__global__ void k_v_transpose(const float* __restrict__ Vf,
                              __bf16* __restrict__ Vt) {
    int i   = blockIdx.x * 256 + threadIdx.x;   // 0 .. 4194303
    int row = i >> 10, col = i & 1023;
    int b = row >> 11, s = row & 2047;
    int kh = col >> 7, d = col & 127;
    Vt[(((size_t)b * 8 + kh) * 128 + d) * 2048 + s] = (__bf16)Vf[i];
}

// ---------------------------------------------------------------------------
// bf16 GEMM: C(MxN,f32) = A(MxK,bf16 rowmajor) * Bt(NxK,bf16 rowmajor)^T
// One wave -> 16M x 64N (4 accumulators); 4 waves/block -> 64M x 64N.
// The shared 64x32 B-tile is staged into LDS with CDNA5 async global->LDS
// copies, double-buffered: DMA of tile i+1 overlaps WMMA on tile i.
// ---------------------------------------------------------------------------
__global__ void k_gemm_bf16(const __bf16* __restrict__ A,
                            const __bf16* __restrict__ Bt,
                            float* __restrict__ C,
                            int M, int N, int K) {
    const int tid  = threadIdx.x;
    const int lane = tid & 31;
    const int wave = tid >> 5;
    const int half = lane >> 4;     // 0: lanes 0-15, 1: lanes 16-31
    const int l16  = lane & 15;
    const int m0   = blockIdx.y * 64 + wave * 16;
    const int n0   = blockIdx.x * 64;

    // double-buffered B tile: 64 N-rows x 32 K of bf16 (4 KB per buffer)
    __shared__ __attribute__((aligned(16))) __bf16 btile[2][64][32];

    const v8f vzero = {0.f, 0.f, 0.f, 0.f, 0.f, 0.f, 0.f, 0.f};
    v8f acc[4];
#pragma unroll
    for (int j = 0; j < 4; ++j) acc[j] = vzero;

    // thread -> two 16B chunks of the 4KB tile: chunk c covers
    // (row = c>>2, elems (c&3)*8 .. +7)
    const int c0row = tid >> 2,        c0off = (tid & 3) * 8;
    const int c1row = (tid + 128) >> 2, c1off = ((tid + 128) & 3) * 8;

    const int rounds = K >> 5;

    // prologue: stage tile 0
    cp_async16(Bt + (size_t)(n0 + c0row) * K + 0 + c0off, &btile[0][c0row][c0off]);
    cp_async16(Bt + (size_t)(n0 + c1row) * K + 0 + c1off, &btile[0][c1row][c1off]);

    for (int i = 0; i < rounds; ++i) {
        const int k0 = i << 5;
        wait_async0();          // stage(i) complete (per-wave, in-order)
        __syncthreads();        // tile i visible to all waves; buf (i+1)&1 free

        if (i + 1 < rounds) {   // kick off stage(i+1), overlapped with WMMA
            const int kn = k0 + 32;
            cp_async16(Bt + (size_t)(n0 + c0row) * K + kn + c0off,
                       &btile[(i + 1) & 1][c0row][c0off]);
            cp_async16(Bt + (size_t)(n0 + c1row) * K + kn + c1off,
                       &btile[(i + 1) & 1][c1row][c1off]);
        }

        // A fragment: lane holds row m0+l16; elems 0-7 -> K=base..base+7,
        // elems 8-15 -> K=base+16..base+23, base = 8*half (ISA 16-bit A layout)
        FragBF a;
        const __bf16* ap = A + (size_t)(m0 + l16) * K + k0 + half * 8;
        a.q[0] = *(const u32x4*)(ap);
        a.q[1] = *(const u32x4*)(ap + 16);

#pragma unroll
        for (int j = 0; j < 4; ++j) {
            // B fragment from LDS: lane holds col j*16+l16; same K layout as A
            FragBF b;
            const __bf16* bl = &btile[i & 1][j * 16 + l16][half * 8];
            b.q[0] = *(const u32x4*)(bl);
            b.q[1] = *(const u32x4*)(bl + 16);
            acc[j] = wmma_bf16(a.v, b.v, acc[j]);
        }
    }

    // C/D layout: VGPR r, lane -> (M = r + 8*half, N = l16)
#pragma unroll
    for (int j = 0; j < 4; ++j)
#pragma unroll
        for (int r = 0; r < 8; ++r)
            C[(size_t)(m0 + r + 8 * half) * N + n0 + j * 16 + l16] = acc[j][r];
}

// ---------------------------------------------------------------------------
// RMSNorm (over full ncols) + RoPE (per 128-dim head), fp32 -> bf16
// out layout: [b][head][s][128]
// ---------------------------------------------------------------------------
__global__ void k_rmsnorm_rope(const float* __restrict__ in,
                               const float* __restrict__ alpha,
                               const float* __restrict__ sin_t,
                               const float* __restrict__ cos_t,
                               __bf16* __restrict__ out,
                               int ncols, int nheads) {
    const int S = 2048;
    int row = blockIdx.x;
    int b = row >> 11, s = row & 2047;
    int tid = threadIdx.x;
    __shared__ float red[256];
    __shared__ float buf[2048];

    float ss = 0.f;
    for (int i = tid; i < ncols; i += 256) {
        float v = in[(size_t)row * ncols + i];
        buf[i] = v;
        ss += v * v;
    }
    red[tid] = ss;
    __syncthreads();
    for (int off = 128; off > 0; off >>= 1) {
        if (tid < off) red[tid] += red[tid + off];
        __syncthreads();
    }
    float rms = rsqrtf(red[0] / (float)ncols + 1e-6f);

    for (int i = tid; i < ncols; i += 256) {
        int h = i >> 7, d = i & 127;
        float xn = alpha[i] * buf[i] * rms;
        int p = (d < 64) ? (i + 64) : (i - 64);
        float xp = alpha[p] * buf[p] * rms;
        float rot = (d < 64) ? -xp : xp;
        float o = xn * cos_t[s * 128 + d] + rot * sin_t[s * 128 + d];
        out[(((size_t)b * nheads + h) * S + s) * 128 + d] = (__bf16)o;
    }
}

// ---------------------------------------------------------------------------
// Causal GQA flash attention. One wave per 16-row Q tile; online softmax over
// 32-key steps; 8 fp32 accumulators (16x128 output per wave).
// Qh: [b][16][s][128] bf16, Kh: [b][8][s][128] bf16, Vt: [b][8][128][s] bf16
// O:  [b*s][2048] bf16 (col = h*128 + d)  -- feeds the O-projection GEMM
// ---------------------------------------------------------------------------
__global__ void k_flash_attn(const __bf16* __restrict__ Qh,
                             const __bf16* __restrict__ Kh,
                             const __bf16* __restrict__ Vt,
                             __bf16* __restrict__ O) {
    const int S = 2048;
    const int lane = threadIdx.x & 31;
    const int wave = threadIdx.x >> 5;
    const int half = lane >> 4;
    const int l16  = lane & 15;

    const int bh  = blockIdx.x;            // 0..31
    const int b   = bh >> 4;
    const int h   = bh & 15;
    const int kvh = h >> 1;                // GQA: 16 q-heads over 8 kv-heads
    const int q0  = (blockIdx.y * 4 + wave) * 16;

    const __bf16* Qp = Qh + ((size_t)(b * 16 + h)  * S) * 128;
    const __bf16* Kp = Kh + ((size_t)(b * 8 + kvh) * S) * 128;
    const __bf16* Vp = Vt + ((size_t)(b * 8 + kvh) * 128) * S;

    __shared__ __attribute__((aligned(16))) __bf16 plds[4][16][32];

    // Q A-fragments: 16x128 split into 4 chunks of K=32
    FragBF qf[4];
#pragma unroll
    for (int c = 0; c < 4; ++c) {
        const __bf16* qp = Qp + (size_t)(q0 + l16) * 128 + c * 32 + half * 8;
        qf[c].q[0] = *(const u32x4*)(qp);
        qf[c].q[1] = *(const u32x4*)(qp + 16);
    }

    const v8f vzero = {0.f, 0.f, 0.f, 0.f, 0.f, 0.f, 0.f, 0.f};
    v8f acc[8];
#pragma unroll
    for (int nd = 0; nd < 8; ++nd) acc[nd] = vzero;
    float m[8], l[8];
#pragma unroll
    for (int r = 0; r < 8; ++r) { m[r] = -1e30f; l[r] = 0.f; }

    const float scale = 0.08838834764831845f;  // 1/sqrt(128)

    for (int jb = 0; jb <= q0 + 15; jb += 32) {
        // ---- scores: S0 = Q * K[jb..jb+15]^T, S1 = Q * K[jb+16..jb+31]^T
        v8f s0 = vzero, s1 = vzero;
#pragma unroll
        for (int c = 0; c < 4; ++c) {
            FragBF bk;
            const __bf16* kp0 = Kp + (size_t)(jb + l16) * 128 + c * 32 + half * 8;
            bk.q[0] = *(const u32x4*)(kp0);
            bk.q[1] = *(const u32x4*)(kp0 + 16);
            s0 = wmma_bf16(qf[c].v, bk.v, s0);
            const __bf16* kp1 = Kp + (size_t)(jb + 16 + l16) * 128 + c * 32 + half * 8;
            bk.q[0] = *(const u32x4*)(kp1);
            bk.q[1] = *(const u32x4*)(kp1 + 16);
            s1 = wmma_bf16(qf[c].v, bk.v, s1);
        }

        // ---- scale + causal mask + online softmax (fp32)
        const int kj0 = jb + l16, kj1 = jb + 16 + l16;
        float p0[8], p1[8];
#pragma unroll
        for (int r = 0; r < 8; ++r) {
            const int qi = q0 + r + 8 * half;
            float v0 = (kj0 <= qi) ? s0[r] * scale : -1e30f;
            float v1 = (kj1 <= qi) ? s1[r] * scale : -1e30f;
            float mx = hmax16(fmaxf(v0, v1));
            float mn = fmaxf(m[r], mx);
            float f  = __expf(m[r] - mn);
            float e0 = __expf(v0 - mn);
            float e1 = __expf(v1 - mn);
            float rs = hsum16(e0 + e1);
            l[r] = l[r] * f + rs;
            m[r] = mn;
            p0[r] = e0;
            p1[r] = e1;
#pragma unroll
            for (int nd = 0; nd < 8; ++nd) acc[nd][r] *= f;
        }

        // ---- transpose P (C layout -> A-fragment layout) through LDS
#pragma unroll
        for (int r = 0; r < 8; ++r) {
            const int mr = r + 8 * half;
            plds[wave][mr][l16]      = (__bf16)p0[r];
            plds[wave][mr][16 + l16] = (__bf16)p1[r];
        }
        asm volatile("s_wait_dscnt 0" ::: "memory");  // DS wave-ordered; data return
        FragBF pf;
        pf.q[0] = *(const u32x4*)&plds[wave][l16][half * 8];
        pf.q[1] = *(const u32x4*)&plds[wave][l16][half * 8 + 16];

        // ---- O += P(16x32) * V(32x128); V^T layout gives contiguous B frags
#pragma unroll
        for (int nd = 0; nd < 8; ++nd) {
            FragBF bv;
            const __bf16* vp = Vp + (size_t)(nd * 16 + l16) * S + jb + half * 8;
            bv.q[0] = *(const u32x4*)(vp);
            bv.q[1] = *(const u32x4*)(vp + 16);
            acc[nd] = wmma_bf16(pf.v, bv.v, acc[nd]);
        }
    }

    // ---- epilogue: normalize by l, write bf16 head-major row for O-proj GEMM
#pragma unroll
    for (int nd = 0; nd < 8; ++nd)
#pragma unroll
        for (int r = 0; r < 8; ++r) {
            float val = acc[nd][r] / l[r];
            size_t orow = (size_t)b * S + q0 + r + 8 * half;
            O[orow * 2048 + h * 128 + nd * 16 + l16] = (__bf16)val;
        }
}

// ---------------------------------------------------------------------------
// Host launcher
// ---------------------------------------------------------------------------
extern "C" void kernel_launch(void* const* d_in, const int* in_sizes, int n_in,
                              void* d_out, int out_size, void* d_ws, size_t ws_size,
                              hipStream_t stream) {
    const float* x       = (const float*)d_in[0];
    const float* sin_t   = (const float*)d_in[1];
    const float* cos_t   = (const float*)d_in[2];
    const float* w_q     = (const float*)d_in[3];
    const float* w_k     = (const float*)d_in[4];
    const float* w_v     = (const float*)d_in[5];
    const float* w_o     = (const float*)d_in[6];
    const float* alpha_q = (const float*)d_in[7];
    const float* alpha_k = (const float*)d_in[8];
    float* out = (float*)d_out;

    // workspace layout (bytes)
    char* ws = (char*)d_ws;
    constexpr size_t OFF_XB   = 0;                       //  8 MB x bf16 (4096x1024)
    constexpr size_t OFF_WTQ  = OFF_XB   +  8388608;     //  4 MB w_q^T (2048x1024)
    constexpr size_t OFF_WTK  = OFF_WTQ  +  4194304;     //  2 MB w_k^T (1024x1024)
    constexpr size_t OFF_WTV  = OFF_WTK  +  2097152;     //  2 MB w_v^T (1024x1024)
    constexpr size_t OFF_WTO  = OFF_WTV  +  2097152;     //  4 MB w_o^T (1024x2048)
    constexpr size_t OFF_QF   = OFF_WTO  +  4194304;     // 32 MB Q fp32 (4096x2048)
    constexpr size_t OFF_KF   = OFF_QF   + 33554432;     // 16 MB K fp32 (4096x1024)
    constexpr size_t OFF_VF   = OFF_KF   + 16777216;     // 16 MB V fp32 (4096x1024)
    constexpr size_t OFF_QH   = OFF_VF   + 16777216;     // 16 MB Q bf16 head-major
    constexpr size_t OFF_KH   = OFF_QH   + 16777216;     //  8 MB K bf16 head-major
    constexpr size_t OFF_VT   = OFF_KH   +  8388608;     //  8 MB V^T bf16
    constexpr size_t OFF_ATTN = OFF_VT   +  8388608;     // 16 MB attn bf16 (4096x2048)

    __bf16* xb    = (__bf16*)(ws + OFF_XB);
    __bf16* wtq   = (__bf16*)(ws + OFF_WTQ);
    __bf16* wtk   = (__bf16*)(ws + OFF_WTK);
    __bf16* wtv   = (__bf16*)(ws + OFF_WTV);
    __bf16* wto   = (__bf16*)(ws + OFF_WTO);
    float*  Qf    = (float*) (ws + OFF_QF);
    float*  Kf    = (float*) (ws + OFF_KF);
    float*  Vf    = (float*) (ws + OFF_VF);
    __bf16* Qh    = (__bf16*)(ws + OFF_QH);
    __bf16* Kh    = (__bf16*)(ws + OFF_KH);
    __bf16* Vt    = (__bf16*)(ws + OFF_VT);
    __bf16* attnb = (__bf16*)(ws + OFF_ATTN);

    // 1) precision conversion / weight transposes
    k_f32_to_bf16  <<<16384, 256, 0, stream>>>(x, xb, 4096 * 1024);
    k_transpose_bf16<<< 8192, 256, 0, stream>>>(w_q, wtq, 1024, 2048);
    k_transpose_bf16<<< 4096, 256, 0, stream>>>(w_k, wtk, 1024, 1024);
    k_transpose_bf16<<< 4096, 256, 0, stream>>>(w_v, wtv, 1024, 1024);
    k_transpose_bf16<<< 8192, 256, 0, stream>>>(w_o, wto, 2048, 1024);

    // 2) projections (WMMA bf16, async-LDS staged B, fp32 out)
    k_gemm_bf16<<<dim3(32, 64), 128, 0, stream>>>(xb, wtq, Qf, 4096, 2048, 1024);
    k_gemm_bf16<<<dim3(16, 64), 128, 0, stream>>>(xb, wtk, Kf, 4096, 1024, 1024);
    k_gemm_bf16<<<dim3(16, 64), 128, 0, stream>>>(xb, wtv, Vf, 4096, 1024, 1024);

    // 3) RMSNorm + RoPE (fp32 math) -> bf16 head-major; V -> bf16 d-major
    k_rmsnorm_rope<<<4096, 256, 0, stream>>>(Qf, alpha_q, sin_t, cos_t, Qh, 2048, 16);
    k_rmsnorm_rope<<<4096, 256, 0, stream>>>(Kf, alpha_k, sin_t, cos_t, Kh, 1024, 8);
    k_v_transpose <<<16384, 256, 0, stream>>>(Vf, Vt);

    // 4) causal GQA flash attention (WMMA bf16, fp32 softmax/acc)
    k_flash_attn<<<dim3(32, 32), 128, 0, stream>>>(Qh, Kh, Vt, attnb);

    // 5) output projection (WMMA bf16, fp32 out)
    k_gemm_bf16<<<dim3(16, 64), 128, 0, stream>>>(attnb, wto, out, 4096, 1024, 2048);
}
